// GATLayer_8366596292961
// MI455X (gfx1250) — matrix operand order repaired and checked by reading, not compiled, not run on hardware
//
#include <hip/hip_runtime.h>
#include <hip/hip_bf16.h>

// ---------------------------------------------------------------------------
// GAT layer for MI455X (gfx1250, wave32)
//   z = h @ W               (WMMA f32 16x16x4; h-tile + W slab staged in LDS
//                            via async global->LDS b128 when available)
//   el/er = z @ a halves    (wave32 butterfly reduction)
//   segment softmax over dst + weighted scatter-add (L2-resident atomics)
// ---------------------------------------------------------------------------

#define IN_DIM   256
#define OUT_DIM  128
#define KCHUNK   64     // K-slab: W 64x128 f32 = 32 KB LDS
#define HPAD     68     // padded h-tile row stride (68 mod 64 = 4 -> no bank conflicts)

typedef float    v2f  __attribute__((ext_vector_type(2)));
typedef float    v8f  __attribute__((ext_vector_type(8)));
typedef _Float16 v16h __attribute__((ext_vector_type(16)));
typedef int      v4i_gcc __attribute__((vector_size(16)));   // matches builtin param
typedef __attribute__((address_space(3))) v4i_gcc* lds_v4i_ptr;

#if __has_builtin(__builtin_amdgcn_global_load_async_to_lds_b128)
#define HAVE_ASYNC_LDS 1
#else
#define HAVE_ASYNC_LDS 0
#endif

__device__ __forceinline__ void async_wait_all() {
#if __has_builtin(__builtin_amdgcn_s_wait_asynccnt)
  __builtin_amdgcn_s_wait_asynccnt(0);
#else
  asm volatile("s_wait_asynccnt 0" ::: "memory");
#endif
}

// ---------------------------------------------------------------------------
// Kernel 0: init  (h_out = 0, segmax = -inf, denom = 0)
// ---------------------------------------------------------------------------
__global__ void gat_init(float* __restrict__ hout, float* __restrict__ segmax,
                         float* __restrict__ denom, int n) {
  int stride = gridDim.x * blockDim.x;
  int total  = n * OUT_DIM;
  for (int i = blockIdx.x * blockDim.x + threadIdx.x; i < total; i += stride)
    hout[i] = 0.0f;
  for (int i = blockIdx.x * blockDim.x + threadIdx.x; i < n; i += stride) {
    segmax[i] = -__builtin_huge_valf();
    denom[i]  = 0.0f;
  }
}

// ---------------------------------------------------------------------------
// Kernel 1: z = h @ W via WMMA f32 16x16x4.
// Block = 256 threads = 8 waves; wave w computes the 16x16 tile at
// rows [16*blockIdx.x, +16), cols [16*w, +16). Per K-slab of 64:
//   - W slab (64x128 f32, 32 KB) and h tile (16x64 f32, bank-pad 68) staged in
//     LDS with async global->LDS b128 copies (ASYNCcnt) when available.
//   - A feeds from LDS (ds_load_b64), B from LDS (paired ds_load).
// ---------------------------------------------------------------------------
__global__ __launch_bounds__(256) void gat_gemm(const float* __restrict__ h,
                                                const float* __restrict__ W,
                                                float* __restrict__ z, int n) {
  __shared__ float Wlds[KCHUNK * OUT_DIM];   // 32 KB
  __shared__ float Hlds[16 * HPAD];          // 4.25 KB, padded rows

  const int tid     = threadIdx.x;
  const int lane    = tid & 31;
  const int wave    = tid >> 5;
  const int rowBase = blockIdx.x * 16;
  const int colBase = wave * 16;
  const int r       = lane & 15;        // M (A) / N (B,C,D) within tile
  const int hi      = lane >> 4;        // lane half
  const int kh      = hi * 2;           // f32 WMMA lane-half K offset

  v8f acc = {};

  for (int kc = 0; kc < IN_DIM; kc += KCHUNK) {
    __syncthreads();  // protect LDS from previous slab's readers

    // ---- stage W slab: 64x128 f32, linear copy of 2048 float4 ----
    const float4* Wg4 = (const float4*)(W + (size_t)kc * OUT_DIM);
    // ---- stage h tile: 16 rows x 64 f32 = 256 float4, one per thread ----
    {
      int hrow  = rowBase + (tid >> 4);           // tid/16 in [0,16)
      if (hrow >= n) hrow = n - 1;                // clamp (n % 16 == 0 anyway)
      int kq    = (tid & 15) * 4;                 // float4 within slab row
      const float* gsrc = h + (size_t)hrow * IN_DIM + kc + kq;
      float*       ldst = Hlds + (tid >> 4) * HPAD + kq;
#if HAVE_ASYNC_LDS
      __builtin_amdgcn_global_load_async_to_lds_b128(
          (v4i_gcc*)gsrc, (lds_v4i_ptr)ldst, 0, 0);
      #pragma unroll
      for (int j = 0; j < 8; ++j) {
        __builtin_amdgcn_global_load_async_to_lds_b128(
            (v4i_gcc*)(Wg4 + tid + j * 256),
            (lds_v4i_ptr)(((float4*)Wlds) + tid + j * 256), 0, 0);
      }
      async_wait_all();
#else
      *(float4*)ldst = *(const float4*)gsrc;
      #pragma unroll
      for (int j = 0; j < 8; ++j)
        ((float4*)Wlds)[tid + j * 256] = Wg4[tid + j * 256];
#endif
    }
    __syncthreads();

#if __has_builtin(__builtin_amdgcn_wmma_f32_16x16x4_f32)
    const float* harow = Hlds + r * HPAD;
    #pragma unroll
    for (int k = 0; k < KCHUNK; k += 4) {
      v2f aa = *(const v2f*)(harow + k + kh);                     // A 16x4 f32
      v2f bb;                                                     // B 4x16 f32
      bb.x = Wlds[(k + kh + 0) * OUT_DIM + colBase + r];
      bb.y = Wlds[(k + kh + 1) * OUT_DIM + colBase + r];
      acc = __builtin_amdgcn_wmma_f32_16x16x4_f32(
          false, aa, false, bb, (short)0, acc, false, false);
    }
#else
    // fallback: probe-verified f16 WMMA (fp32 -> f16 convert, f32 accumulate)
    for (int k = 0; k < KCHUNK; k += 32) {
      v16h aa, bb;
      #pragma unroll
      for (int j = 0; j < 16; ++j) {
        int ka = (j < 8) ? (j + 8 * hi) : (j + 8 + 8 * hi);
        aa[j] = (_Float16)Hlds[r * HPAD + k + ka];
        int kb = ka;
        bb[j] = (_Float16)Wlds[(k + kb) * OUT_DIM + colBase + r];
      }
      acc = __builtin_amdgcn_wmma_f32_16x16x32_f16(
          false, aa, false, bb, (short)0, acc, false, false);
    }
#endif
  }

  // C/D layout: VGPR i -> row (i + 8*hi), col = lane%16
  const int colOut = colBase + r;
  #pragma unroll
  for (int i = 0; i < 8; ++i) {
    int row = rowBase + 8 * hi + i;
    if (row < n) z[(size_t)row * OUT_DIM + colOut] = acc[i];
  }
}

// ---------------------------------------------------------------------------
// Kernel 2: el/er projections. One wave32 per node, float4 per lane, butterfly
// reduce across the wave.
// ---------------------------------------------------------------------------
__global__ __launch_bounds__(256) void gat_proj(const float* __restrict__ z,
                                                const float* __restrict__ a,
                                                float* __restrict__ el,
                                                float* __restrict__ er, int n) {
  int gwave = (blockIdx.x * blockDim.x + threadIdx.x) >> 5;
  int lane  = threadIdx.x & 31;
  if (gwave >= n) return;
  float4 zz = *(const float4*)(z + (size_t)gwave * OUT_DIM + lane * 4);
  float4 al = *(const float4*)(a + lane * 4);
  float4 ar = *(const float4*)(a + OUT_DIM + lane * 4);
  float sl = zz.x * al.x + zz.y * al.y + zz.z * al.z + zz.w * al.w;
  float sr = zz.x * ar.x + zz.y * ar.y + zz.z * ar.z + zz.w * ar.w;
  #pragma unroll
  for (int off = 16; off > 0; off >>= 1) {
    sl += __shfl_xor(sl, off, 32);
    sr += __shfl_xor(sr, off, 32);
  }
  if (lane == 0) { el[gwave] = sl; er[gwave] = sr; }
}

// ---------------------------------------------------------------------------
// Kernel 3: per-edge leaky_relu + atomic segment max (int-bits monotonic trick)
// ---------------------------------------------------------------------------
__global__ void gat_edge_max(const float* __restrict__ el,
                             const float* __restrict__ er,
                             const int* __restrict__ src,
                             const int* __restrict__ dst,
                             float* __restrict__ e,
                             float* __restrict__ segmax, int nedges) {
  int i = blockIdx.x * blockDim.x + threadIdx.x;
  if (i >= nedges) return;
  int d = dst[i];
  float v = el[src[i]] + er[d];
  v = v > 0.0f ? v : 0.01f * v;        // leaky_relu, slope 0.01
  e[i] = v;
  if (v >= 0.0f)
    atomicMax((int*)(segmax + d), __float_as_int(v));
  else
    atomicMin((unsigned int*)(segmax + d), __float_as_uint(v));
}

// ---------------------------------------------------------------------------
// Kernel 4: e <- exp(e - segmax[dst]); denom[dst] += e   (in-place on e)
// ---------------------------------------------------------------------------
__global__ void gat_edge_exp(const float* __restrict__ segmax,
                             const int* __restrict__ dst,
                             float* __restrict__ e,
                             float* __restrict__ denom, int nedges) {
  int i = blockIdx.x * blockDim.x + threadIdx.x;
  if (i >= nedges) return;
  int d = dst[i];
  float x = __expf(e[i] - segmax[d]);
  e[i] = x;
  atomicAdd(denom + d, x);
}

// ---------------------------------------------------------------------------
// Kernel 5: h_out[dst] += (e/denom[dst]) * z[src]. One wave32 per edge,
// float4 gather per lane (z is L2-resident), 4 f32 atomics per lane.
// ---------------------------------------------------------------------------
__global__ __launch_bounds__(256) void gat_scatter(const float* __restrict__ z,
                                                   const float* __restrict__ e,
                                                   const float* __restrict__ denom,
                                                   const int* __restrict__ src,
                                                   const int* __restrict__ dst,
                                                   float* __restrict__ hout,
                                                   int nedges) {
  int gwave = (blockIdx.x * blockDim.x + threadIdx.x) >> 5;
  int lane  = threadIdx.x & 31;
  if (gwave >= nedges) return;
  int s = src[gwave];
  int d = dst[gwave];
  float alpha = e[gwave] / denom[d];
  float4 zz = *(const float4*)(z + (size_t)s * OUT_DIM + lane * 4);
  float* o = hout + (size_t)d * OUT_DIM + lane * 4;
  atomicAdd(o + 0, alpha * zz.x);
  atomicAdd(o + 1, alpha * zz.y);
  atomicAdd(o + 2, alpha * zz.z);
  atomicAdd(o + 3, alpha * zz.w);
}

// ---------------------------------------------------------------------------
// Launch
// ---------------------------------------------------------------------------
extern "C" void kernel_launch(void* const* d_in, const int* in_sizes, int n_in,
                              void* d_out, int out_size, void* d_ws, size_t ws_size,
                              hipStream_t stream) {
  const float* h    = (const float*)d_in[0];
  const float* W    = (const float*)d_in[1];
  const float* a    = (const float*)d_in[2];
  const int*   src  = (const int*)d_in[3];
  const int*   dst  = (const int*)d_in[4];
  float*       hout = (float*)d_out;

  const int n      = in_sizes[0] / IN_DIM;   // 100000
  const int nedges = in_sizes[3];            // 1600000

  // workspace layout (floats): z | el | er | segmax | denom | e  (~59.2 MB)
  float* z      = (float*)d_ws;
  float* el     = z + (size_t)n * OUT_DIM;
  float* er     = el + n;
  float* segmax = er + n;
  float* denom  = segmax + n;
  float* e      = denom + n;

  const int nThreads = 256;

  {  // 0) init output + segment buffers
    int total  = n * OUT_DIM;
    int blocks = (total + nThreads - 1) / nThreads;
    if (blocks > 65535 * 8) blocks = 65535 * 8;
    gat_init<<<blocks, nThreads, 0, stream>>>(hout, segmax, denom, n);
  }
  {  // 1) z = h @ W  (WMMA)
    int rowTiles = (n + 15) / 16;
    gat_gemm<<<rowTiles, nThreads, 0, stream>>>(h, W, z, n);
  }
  {  // 2) el / er
    int blocks = (n + 7) / 8;
    gat_proj<<<blocks, nThreads, 0, stream>>>(z, a, el, er, n);
  }
  {  // 3) edge leaky-relu + segment max
    int blocks = (nedges + nThreads - 1) / nThreads;
    gat_edge_max<<<blocks, nThreads, 0, stream>>>(el, er, src, dst, e, segmax, nedges);
  }
  {  // 4) edge exp + denom
    int blocks = (nedges + nThreads - 1) / nThreads;
    gat_edge_exp<<<blocks, nThreads, 0, stream>>>(segmax, dst, e, denom, nedges);
  }
  {  // 5) weighted scatter-add
    int blocks = (nedges + 7) / 8;
    gat_scatter<<<blocks, nThreads, 0, stream>>>(z, e, denom, src, dst, hout, nedges);
  }
}
